// GraphAttentionLayer_38233798869806
// MI455X (gfx1250) — matrix operand order repaired and checked by reading, not compiled, not run on hardware
//
#include <hip/hip_runtime.h>
#include <hip/hip_bf16.h>

// MI455X / gfx1250, wave32. Attention collapses to adj/deg row normalization
// (softmax rows are constant over unmasked entries), so out = (adj/deg) @ h
// with h = feat@Wf^T + bf + h_mm. Both GEMMs use v_wmma_f32_16x16x32_f16.
// adj (268MB, read-once, > 192MB L2) streams with TH=NT so h16T (16MB) stays
// L2-resident for kernel C's A-operand re-reads.

typedef _Float16 half_t;
typedef __attribute__((ext_vector_type(16))) _Float16 v16h;
typedef __attribute__((ext_vector_type(8)))  _Float16 v8h;
typedef __attribute__((ext_vector_type(8)))  float    v8f;
typedef __attribute__((ext_vector_type(4)))  float    v4f;
typedef __attribute__((ext_vector_type(4)))  int      v4i;

#define B_  16
#define N_  2048
#define F_  256
#define C_  768
#define HW_ 196

static __device__ __forceinline__ v16h mk16(v8h lo, v8h hi) {
  v16h r;
#pragma unroll
  for (int i = 0; i < 8; ++i) { r[i] = lo[i]; r[i + 8] = hi[i]; }
  return r;
}

// ---------------------------------------------------------------------------
// Kernel A: global-avg-pool + h_mm = pool@Wm^T + bm ; zero s ; Wf -> f16
// grid = 16 (one block per batch), 256 threads
// ---------------------------------------------------------------------------
__global__ __launch_bounds__(256)
void prep_kernel(const float* __restrict__ mm, const float* __restrict__ Wm,
                 const float* __restrict__ bm, const float* __restrict__ Wf,
                 float* __restrict__ hmm, float* __restrict__ sbuf,
                 half_t* __restrict__ wf16) {
  __shared__ float pooled[C_];
  const int b = blockIdx.x, t = threadIdx.x;
  for (int c = t; c < C_; c += 256) {
    const float* p = mm + ((size_t)(b * C_ + c)) * HW_;
    float s = 0.0f;
#pragma unroll 4
    for (int i = 0; i < HW_; ++i) s += p[i];
    pooled[c] = s * (1.0f / (float)HW_);
  }
  sbuf[b * F_ + t] = 0.0f;                 // zero s accumulator each launch
#pragma unroll
  for (int r = 0; r < 16; ++r) {           // Wf -> f16 (this block's chunk)
    const int idx = b * 4096 + r * 256 + t;
    wf16[idx] = (half_t)Wf[idx];
  }
  __syncthreads();
  const float* wrow = Wm + (size_t)t * C_;
  float acc = bm[t];
#pragma unroll 4
  for (int c = 0; c < C_; ++c) acc += pooled[c] * wrow[c];
  hmm[b * F_ + t] = acc;
}

// ---------------------------------------------------------------------------
// Kernel B: h = feat(32768x256) @ Wf^T + bf + hmm   via WMMA f16->f32
// Wave: 64 rows(i) x 32 cols(f) = 4x2 accum tiles; no LDS.
//  A (features, row-major [i][k]) -> direct 8-split A-frag loads (NT, f32->f16)
//  B (Wf16, row-major [f][k])     -> direct 8-split B-frag loads (L2-hot)
// Writes h16T[b][f][i] (f16, 16B packed) + atomic s[b][f] partials.
// grid = 512 blocks x 256 thr (4096 waves = 512 i-bands x 8 f-bands)
// ---------------------------------------------------------------------------
__global__ __launch_bounds__(256)
void hgemm_kernel(const float* __restrict__ feat, const half_t* __restrict__ wf16,
                  const float* __restrict__ bf, const float* __restrict__ hmm,
                  half_t* __restrict__ h16T, float* __restrict__ sbuf) {
  const int tid   = threadIdx.x;
  const int lane  = tid & 31;
  const int wid   = blockIdx.x * 8 + (tid >> 5);
  const int iband = wid >> 3;              // 0..511 (64 global rows each)
  const int fband = wid & 7;               // 0..7   (32 f cols each)
  const int rowBase = iband * 64;          // global row in [0,32768)
  const int fBase   = fband * 32;
  const int b       = rowBase >> 11;       // batch (64 | 2048, no crossing)
  const int laneM   = lane & 15;
  const int laneH   = lane >> 4;

  v8f acc[4][2] = {};

  for (int k0 = 0; k0 < F_; k0 += 32) {
    const int kbA = k0 + 8 * laneH;        // A-frag 8-split K base
    v16h afrag[4];
#pragma unroll
    for (int mt = 0; mt < 4; ++mt) {
      const float* arow = feat + ((size_t)(rowBase + mt * 16 + laneM)) * F_ + kbA;
      const v4f f0 = __builtin_nontemporal_load((const v4f*)(arow));
      const v4f f1 = __builtin_nontemporal_load((const v4f*)(arow + 4));
      const v4f f2 = __builtin_nontemporal_load((const v4f*)(arow + 16));
      const v4f f3 = __builtin_nontemporal_load((const v4f*)(arow + 20));
      v8h lo = {(half_t)f0.x, (half_t)f0.y, (half_t)f0.z, (half_t)f0.w,
                (half_t)f1.x, (half_t)f1.y, (half_t)f1.z, (half_t)f1.w};
      v8h hi = {(half_t)f2.x, (half_t)f2.y, (half_t)f2.z, (half_t)f2.w,
                (half_t)f3.x, (half_t)f3.y, (half_t)f3.z, (half_t)f3.w};
      afrag[mt] = mk16(lo, hi);
    }
    const int kbB = k0 + 8 * laneH;        // B-frag 8-split K base
#pragma unroll
    for (int nt = 0; nt < 2; ++nt) {
      const half_t* brow = wf16 + ((size_t)(fBase + nt * 16 + laneM)) * F_ + kbB;
      const v16h bfrag = mk16(*(const v8h*)brow, *(const v8h*)(brow + 16));
#pragma unroll
      for (int mt = 0; mt < 4; ++mt)
        acc[mt][nt] = __builtin_amdgcn_wmma_f32_16x16x32_f16(
            false, afrag[mt], false, bfrag, (short)0, acc[mt][nt], false, false);
    }
  }

  // Epilogue: add bias + h_mm, store h16T, accumulate s = sum_i h.
#pragma unroll
  for (int nt = 0; nt < 2; ++nt) {
    const int f = fBase + nt * 16 + laneM;
    const float bias = bf[f] + hmm[b * F_ + f];
    float ssum = 0.0f;
#pragma unroll
    for (int mt = 0; mt < 4; ++mt) {
      v8h pk;
#pragma unroll
      for (int r = 0; r < 8; ++r) {
        const float v = acc[mt][nt][r] + bias;
        ssum += v;
        pk[r] = (half_t)v;
      }
      const int iBase = (rowBase & (N_ - 1)) + mt * 16 + 8 * laneH;
      *(v8h*)(h16T + ((size_t)(b * F_ + f)) * N_ + iBase) = pk;  // 16B, RT
    }
    ssum += __shfl_xor(ssum, 16, 32);       // lanes L and L+16 share f
    if (laneH == 0) atomicAdd(&sbuf[b * F_ + f], ssum);
  }
}

// ---------------------------------------------------------------------------
// Kernel C: out^T[f][i] = h^T @ adj^T, then scale rows by 1/deg (f32).
// Block: full 256 f x 64 i, K loop over j. adj tile (64x32 i32) streamed NT
// from HBM exactly once, converted to f16 into double-buffered LDS (1 barrier
// per K-step); per-row degree folded into staging. 8 waves = 4(f) x 2(i).
// grid = 512 blocks (16 batches x 32 i-bands) x 256 thr
// ---------------------------------------------------------------------------
__global__ __launch_bounds__(256)
void attn_kernel(const int* __restrict__ adj, const half_t* __restrict__ h16T,
                 const float* __restrict__ sbuf, float* __restrict__ out) {
  __shared__ half_t lds_bh[2][64 * 32];
  __shared__ float  lds_deg[64];

  const int tid  = threadIdx.x;
  const int lane = tid & 31;
  const int wav  = tid >> 5;
  const int wm   = wav >> 1;               // 0..3 : 64-wide f band
  const int wn   = wav & 1;                // 0..1 : 32-wide i sub-band
  const int b    = blockIdx.x >> 5;
  const int i0   = (blockIdx.x & 31) * 64;
  const int laneM = lane & 15;
  const int laneH = lane >> 4;

  // staging: thread t handles 8 adj ints of row (t>>2), cols (t&3)*8..+7
  const int sRow = tid >> 2;
  const int sCol = (tid & 3) * 8;
  const int* adjRow = adj + ((size_t)b * N_ + (size_t)(i0 + sRow)) * N_ + sCol;
  int cnt = 0;

  if (tid < 64) lds_deg[tid] = 0.0f;

  // Prologue: stage tile k=0 into buffer 0 (NT stream loads).
  {
    const v4i a0 = __builtin_nontemporal_load((const v4i*)(adjRow));
    const v4i a1 = __builtin_nontemporal_load((const v4i*)(adjRow + 4));
    cnt += a0.x + a0.y + a0.z + a0.w + a1.x + a1.y + a1.z + a1.w;
    v8h pk = {(half_t)a0.x, (half_t)a0.y, (half_t)a0.z, (half_t)a0.w,
              (half_t)a1.x, (half_t)a1.y, (half_t)a1.z, (half_t)a1.w};
    *(v8h*)(&lds_bh[0][sRow * 32 + sCol]) = pk;
  }
  __syncthreads();

  v8f acc[4][2] = {};

  for (int k0 = 0; k0 < N_; k0 += 32) {
    const int  cur  = (k0 >> 5) & 1;
    const bool more = (k0 + 32) < N_;

    // Issue next tile's stream loads early (hide behind WMMAs below).
    v4i a0 = {}, a1 = {};
    if (more) {
      a0 = __builtin_nontemporal_load((const v4i*)(adjRow + k0 + 32));
      a1 = __builtin_nontemporal_load((const v4i*)(adjRow + k0 + 36));
    }

    // B-frags (adj^T): lane = i, K 8-split per lane-half, from LDS[cur].
    v16h bfrag[2];
#pragma unroll
    for (int nt = 0; nt < 2; ++nt) {
      const half_t* p = &lds_bh[cur][(wn * 32 + nt * 16 + laneM) * 32 + 8 * laneH];
      bfrag[nt] = mk16(*(const v8h*)p, *(const v8h*)(p + 16));
    }
    // A-frags (h^T): straight from h16T (kept L2-hot by NT adj stream).
    const int kbA = k0 + 8 * laneH;
#pragma unroll
    for (int mt = 0; mt < 4; ++mt) {
      const half_t* arow =
          h16T + ((size_t)(b * F_ + wm * 64 + mt * 16 + laneM)) * N_ + kbA;
      const v16h af = mk16(*(const v8h*)arow, *(const v8h*)(arow + 16));
#pragma unroll
      for (int nt = 0; nt < 2; ++nt)
        acc[mt][nt] = __builtin_amdgcn_wmma_f32_16x16x32_f16(
            false, af, false, bfrag[nt], (short)0, acc[mt][nt], false, false);
    }

    // Stage next tile into the other buffer (uniform branch; safe w/ barrier).
    if (more) {
      cnt += a0.x + a0.y + a0.z + a0.w + a1.x + a1.y + a1.z + a1.w;
      v8h pk = {(half_t)a0.x, (half_t)a0.y, (half_t)a0.z, (half_t)a0.w,
                (half_t)a1.x, (half_t)a1.y, (half_t)a1.z, (half_t)a1.w};
      *(v8h*)(&lds_bh[cur ^ 1][sRow * 32 + sCol]) = pk;
    }
    __syncthreads();                       // one barrier per K-step
  }

  atomicAdd(&lds_deg[sRow], (float)cnt);   // 4 partials per row
  __syncthreads();

  // Epilogue: out[b][i][f] = acc * (1/deg_i); deg==0 -> s[b][f]/N fallback.
#pragma unroll
  for (int nt = 0; nt < 2; ++nt) {
    const int iRel = wn * 32 + nt * 16 + laneM;
    const float deg = lds_deg[iRel];
    const float scl = (deg > 0.0f) ? (1.0f / deg) : 0.0f;
    float* orow = out + ((size_t)(b * N_ + i0 + iRel)) * F_;
#pragma unroll
    for (int mt = 0; mt < 4; ++mt) {
      const int fB = wm * 64 + mt * 16 + 8 * laneH;   // 8 consecutive f
      float vals[8];
#pragma unroll
      for (int r = 0; r < 8; ++r) vals[r] = acc[mt][nt][r] * scl;
      if (deg == 0.0f) {
#pragma unroll
        for (int r = 0; r < 8; ++r)
          vals[r] = sbuf[b * F_ + fB + r] * (1.0f / (float)N_);
      }
      v4f lo = {vals[0], vals[1], vals[2], vals[3]};
      v4f hi = {vals[4], vals[5], vals[6], vals[7]};
      __builtin_nontemporal_store(lo, (v4f*)(orow + fB));      // write-once
      __builtin_nontemporal_store(hi, (v4f*)(orow + fB + 4));
    }
  }
}

// ---------------------------------------------------------------------------
extern "C" void kernel_launch(void* const* d_in, const int* in_sizes, int n_in,
                              void* d_out, int out_size, void* d_ws, size_t ws_size,
                              hipStream_t stream) {
  (void)in_sizes; (void)n_in; (void)out_size; (void)ws_size;
  const float* feat = (const float*)d_in[0];   // (16,2048,1,256)
  const int*   adj  = (const int*)d_in[1];     // (16,2048,2048)
  const float* mm   = (const float*)d_in[2];   // (16,768,14,14)
  const float* Wf   = (const float*)d_in[3];   // (256,256)
  const float* bf   = (const float*)d_in[4];   // (256,)
  const float* Wm   = (const float*)d_in[5];   // (256,768)
  const float* bm   = (const float*)d_in[6];   // (256,)
  // d_in[7]=Wa, d_in[8]=ba: mathematically dead (softmax over rows whose
  // unmasked entries are all equal -> att = adj/deg regardless of e).
  float* out = (float*)d_out;

  char*  ws   = (char*)d_ws;                    // ~16.2 MB used
  float* hmm  = (float*)(ws);                   // 16*256 f32   (16 KB)
  float* sbuf = (float*)(ws + 16384);           // 16*256 f32   (16 KB)
  half_t* wf16 = (half_t*)(ws + 32768);         // 256*256 f16  (128 KB)
  half_t* h16T = (half_t*)(ws + 163840);        // 16*256*2048 f16 (16 MB)

  prep_kernel <<<16,  256, 0, stream>>>(mm, Wm, bm, Wf, hmm, sbuf, wf16);
  hgemm_kernel<<<512, 256, 0, stream>>>(feat, wf16, bf, hmm, h16T, sbuf);
  attn_kernel <<<512, 256, 0, stream>>>(adj, h16T, sbuf, out);
}
